// ModelIAS_8478265442317
// MI455X (gfx1250) — matrix operand order repaired and checked by reading, not compiled, not run on hardware
//
#include <hip/hip_runtime.h>
#include <hip/hip_bf16.h>

typedef float v8f  __attribute__((ext_vector_type(8)));
typedef __bf16 v16bf __attribute__((ext_vector_type(16)));
typedef __bf16 v4bf  __attribute__((ext_vector_type(4)));

// ---------------------------------------------------------------------------
// Problem constants (from reference)
// ---------------------------------------------------------------------------
#define BATCH 64
#define SEQ   513
#define TLEN  512
#define HDIM  1024
#define ODIM  128

// Kernel 1 tiling
#define MTILE 128          // t-rows per workgroup
#define KC    64           // K-chunk staged in LDS (2 WMMA K-steps of 32)
#define KCP   72           // LDS row stride in bf16 halves: 144B, 16B-aligned,
                           // conflict-free for strided 16B lane reads

// ---------------------------------------------------------------------------
// Kernel 0: one-time W prep: fp32 (H,O) -> bf16, transposed + chunk-major:
//   wtg[(k/KC)*(ODIM*KC) + o*KC + (k%KC)] = bf16(W[k][o])
// This makes per-chunk staging a linear 16KB copy and every WMMA B-fragment
// (16 consecutive K of one column) two contiguous ds_load_b128's.
// ---------------------------------------------------------------------------
__global__ __launch_bounds__(256) void
wprep_kernel(const float* __restrict__ W, __bf16* __restrict__ wtg)
{
    const int gid = blockIdx.x * 256 + threadIdx.x;   // 0 .. H*O-1
    const int o = gid & (ODIM - 1);
    const int k = gid >> 7;
    const float v = W[(size_t)k * ODIM + o];          // coalesced in o
    wtg[(size_t)(k >> 6) * (ODIM * KC) + o * KC + (k & (KC - 1))] = (__bf16)v;
}

// ---------------------------------------------------------------------------
// Kernel 1: logits[b, t, o] = sum_h x[b, t+1, h] * W[h, o] + bias[o]
// One WG = (batch, 128-row t-tile); 8 waves, wave w owns M-strip w*16 and all
// 8 N-tiles. bf16 WMMA 16x16x32, fp32 accumulate. x converted fp32->bf16 on
// the fly during LDS staging (x is read from HBM exactly once, as fp32).
// ---------------------------------------------------------------------------
__global__ __launch_bounds__(256) void
gemm_bias_kernel(const float*  __restrict__ x,     // (B, S, H) f32
                 const __bf16* __restrict__ wtg,   // chunk-major transposed W
                 const float*  __restrict__ bias,  // (O)
                 float* __restrict__ logits)       // (B, T, O) workspace
{
    const int tt   = blockIdx.x;          // t-tile: 0..3
    const int b    = blockIdx.y;          // batch
    const int tid  = threadIdx.x;
    const int wave = tid >> 5;
    const int lane = tid & 31;
    const int l16  = lane & 15;
    const int hi   = lane >> 4;           // 0: lanes 0-15, 1: lanes 16-31

    __shared__ __bf16 xs[MTILE * KCP];    // x tile  [128][64] padded -> 18 KB
    __shared__ __bf16 wt[ODIM  * KCP];    // W chunk [128 o][64 k]    -> 18 KB

    const int m0 = wave * 16;

    v8f acc[8];
#pragma unroll
    for (int n = 0; n < 8; ++n)
        acc[n] = (v8f){0.f, 0.f, 0.f, 0.f, 0.f, 0.f, 0.f, 0.f};

    // x rows start at sequence position 1 (reference drops hidden_states[:,0])
    const float* xbase = x + ((size_t)(b * SEQ + 1 + tt * MTILE)) * HDIM;

    // staging index precomputes
    const int xrow = tid >> 1;                  // 0..127
    const int xcb  = (tid & 1) * 32;            // col block: 0 or 32

    for (int kc = 0; kc < HDIM; kc += KC) {
        __syncthreads();

        // --- stage x tile: 128 rows x 64 cols fp32 -> bf16.
        // Thread: row xrow, cols xcb..xcb+31 (8 x float4 -> 8 x b64 LDS store).
        {
            const float* src = xbase + (size_t)xrow * HDIM + kc + xcb;
            __bf16* dst = &xs[xrow * KCP + xcb];
#pragma unroll
            for (int j = 0; j < 8; ++j) {
                const float4 v = *(const float4*)(src + j * 4);
                v4bf h;
                h.x = (__bf16)v.x; h.y = (__bf16)v.y;
                h.z = (__bf16)v.z; h.w = (__bf16)v.w;
                *(v4bf*)(dst + j * 4) = h;
            }
        }
        // --- stage W chunk: linear 16KB bf16 copy (fully coalesced b128),
        // un-linearized into the padded LDS rows (col stays inside a row).
        {
            const __bf16* src = wtg + (size_t)(kc >> 6) * (ODIM * KC);
#pragma unroll
            for (int p = 0; p < 8; ++p) {
                const int idx = p * 2048 + tid * 8;          // half index
                const uint4 v = *(const uint4*)(src + idx);  // 16B, linear
                const int r = idx >> 6;                      // idx / 64
                const int c = idx & 63;                      // multiple of 8
                *(uint4*)(&wt[r * KCP + c]) = v;
            }
        }
        __syncthreads();

        // --- 2 K-steps of 32; 8 N-tiles each.
        // A 16x32 bf16 layout: lanes 0-15 hold K 0..15 of row M=lane;
        //                      lanes 16-31 hold K 16..31 of M=lane-16.
        // B 32x16 bf16 layout: lanes 0-15 hold K 0..15 of col N=lane;
        //                      lanes 16-31 hold K 16..31 of N=lane-16.
#pragma unroll
        for (int ks = 0; ks < KC; ks += 32) {
            union { v16bf v; uint4 q[2]; } a;
            const __bf16* ap = &xs[(m0 + l16) * KCP + ks + hi * 16];
            a.q[0] = *(const uint4*)(ap);        // K run  hi*16 .. +7
            a.q[1] = *(const uint4*)(ap + 8);    // K run  hi*16+8 .. +7
#pragma unroll
            for (int n = 0; n < 8; ++n) {
                union { v16bf v; uint4 q[2]; } bb;
                const __bf16* bp = &wt[(n * 16 + l16) * KCP + ks + hi * 16];
                bb.q[0] = *(const uint4*)(bp);       // K 0..7  (+hi*16)
                bb.q[1] = *(const uint4*)(bp + 8);   // K 8..15 (+hi*16)
                acc[n] = __builtin_amdgcn_wmma_f32_16x16x32_bf16(
                    /*neg_a=*/false, a.v, /*neg_b=*/false, bb.v,
                    /*c_mod=*/(short)0, acc[n],
                    /*reuse_a=*/false, /*reuse_b=*/false);
            }
        }
    }

    // --- bias + store. C/D layout: VGPR e -> M = e + 8*hi, N = n*16 + l16.
    float* obase = logits + ((size_t)(b * TLEN + tt * MTILE + m0)) * ODIM;
#pragma unroll
    for (int n = 0; n < 8; ++n) {
        const float bv = bias[n * 16 + l16];
#pragma unroll
        for (int e = 0; e < 8; ++e) {
            const int M = e + hi * 8;
            obase[(size_t)M * ODIM + n * 16 + l16] = acc[n][e] + bv;
        }
    }
}

// ---------------------------------------------------------------------------
// Kernel 2: segment mean over t (contiguous runs) + transpose to (B, O, T).
// One WG per batch, 128 threads (thread = output channel o). Segment structure
// is identical for every o -> no divergence; logits reads are 512B coalesced.
// merged[o][j] lives in 256 KB dynamic LDS (CDNA5 WGP has 320 KB), enabling a
// single-pass conflict-free transposed, coalesced write-out.
// ---------------------------------------------------------------------------
__global__ __launch_bounds__(128) void
segment_mean_transpose_kernel(const float* __restrict__ logits, // (B, T, O)
                              const int*   __restrict__ seg,    // (B, T)
                              float* __restrict__ out)          // (B, O, T)
{
    extern __shared__ float smem[];
    float* merged = smem;                       // [128][512]
    int*   ssh    = (int*)(smem + ODIM * TLEN); // [512]

    const int b = blockIdx.x;
    const int o = threadIdx.x;                  // 0..127

    for (int t = o; t < TLEN; t += ODIM)
        ssh[t] = seg[(size_t)b * TLEN + t];
    __syncthreads();

    const float* lcol = logits + (size_t)b * TLEN * ODIM + o;
    float acc = 0.f;
    int   cnt = 0;
    int   cur = ssh[0];                          // == 0 by construction
    for (int t = 0; t < TLEN; ++t) {
        const int s = ssh[t];
        if (s != cur) {
            merged[(size_t)o * TLEN + cur] = acc / (float)cnt;
            acc = 0.f; cnt = 0; cur = s;
        }
        acc += lcol[(size_t)t * ODIM];
        ++cnt;
    }
    merged[(size_t)o * TLEN + cur] = acc / (float)cnt;
    const int nseg = ssh[TLEN - 1] + 1;          // segments j >= nseg are empty
    __syncthreads();

    // Write-out: consecutive lanes -> consecutive j: LDS conflict-free,
    // global stores fully coalesced. Empty segments -> 0.
    float* ob = out + (size_t)b * ODIM * TLEN;
    for (int idx = o; idx < ODIM * TLEN; idx += ODIM) {
        const int oo = idx >> 9;                 // idx / 512
        const int j  = idx & (TLEN - 1);         // idx % 512
        ob[idx] = (j < nseg) ? merged[(size_t)oo * TLEN + j] : 0.f;
    }
}

// ---------------------------------------------------------------------------
extern "C" void kernel_launch(void* const* d_in, const int* in_sizes, int n_in,
                              void* d_out, int out_size, void* d_ws, size_t ws_size,
                              hipStream_t stream)
{
    const float* hs   = (const float*)d_in[0];  // (64, 513, 1024) f32
    const float* W    = (const float*)d_in[1];  // (1024, 128)     f32
    const float* bias = (const float*)d_in[2];  // (128,)          f32
    const int*   seg  = (const int*)d_in[3];    // (64, 512)       i32

    // workspace: logits (16 MiB f32) then transposed bf16 W (256 KiB)
    float*  logits = (float*)d_ws;
    __bf16* wtg    = (__bf16*)((char*)d_ws
                       + (size_t)BATCH * TLEN * ODIM * sizeof(float));

    wprep_kernel<<<dim3((HDIM * ODIM) / 256), dim3(256), 0, stream>>>(W, wtg);

    dim3 g1(TLEN / MTILE, BATCH);               // (4, 64)
    gemm_bias_kernel<<<g1, dim3(256), 0, stream>>>(hs, wtg, bias, logits);

    const size_t sh2 = (size_t)ODIM * TLEN * sizeof(float)
                     + (size_t)TLEN * sizeof(int);   // 258 KB dynamic LDS
    segment_mean_transpose_kernel<<<dim3(BATCH), dim3(ODIM), sh2, stream>>>(
        logits, seg, (float*)d_out);
}
